// VNRSchedulerAgent_35124242546946
// MI455X (gfx1250) — matrix operand order repaired and checked by reading, not compiled, not run on hardware
//
#include <hip/hip_runtime.h>

#define DEVI __device__ __forceinline__

// ---------------------------------------------------------------------------
// CDNA5 WMMA types (wave32): 16x16x32 bf16 -> f32
// ---------------------------------------------------------------------------
typedef __attribute__((ext_vector_type(16))) __bf16 v16bf;
typedef __attribute__((ext_vector_type(8)))  __bf16 v8bf;
typedef __attribute__((ext_vector_type(8)))  float  v8f;

DEVI __bf16 f2bf(float f) {            // RNE f32->bf16 (used only in prep kernels)
  unsigned u = __float_as_uint(f);
  unsigned r = (u + 0x7FFFu + ((u >> 16) & 1u)) >> 16;
  unsigned short h = (unsigned short)r;
  return __builtin_bit_cast(__bf16, h);
}

DEVI void atomAddF(float* p, float v) {
  unsafeAtomicAdd(p, v);   // native global_atomic_add_f32 (no CAS loop)
}

DEVI void atomMaxF(float* addr, float value) {
  // float max via signed/unsigned integer atomics (valid for mixed signs)
  if (value >= 0.f) atomicMax((int*)addr, __float_as_int(value));
  else              atomicMin((unsigned int*)addr, __float_as_uint(value));
}

// ---------------------------------------------------------------------------
// Prep: f32 -> bf16 elementwise, and W (K x M, f32) -> WT (M x K, bf16).
// Both are one-shot, memory-bound; keeps all conversion out of the GEMM loop.
// ---------------------------------------------------------------------------
__global__ void k_cvt_bf16(const float* __restrict__ src, __bf16* __restrict__ dst, int n) {
  int i = blockIdx.x * blockDim.x + threadIdx.x;
  if (i < n) dst[i] = f2bf(src[i]);
}

__global__ void k_transpose_bf16(const float* __restrict__ W, __bf16* __restrict__ WT,
                                 int K, int M) {
  int idx = blockIdx.x * blockDim.x + threadIdx.x;
  if (idx >= K * M) return;
  int m = idx / K, k = idx - m * K;        // contiguous writes along K
  WT[(size_t)m * K + k] = f2bf(W[(size_t)k * M + m]);
}

// ---------------------------------------------------------------------------
// WMMA GEMM: C[N x M] = A[N x K] * B[K x M]; A in bf16 (N x K row-major),
// B pre-transposed to BT (M x K row-major, bf16), C f32.
// One wave computes a 16x32 strip (two 16x16 tiles sharing the A fragment).
// Fragment layouts used here (contiguous k = 16h + s, same in A and B):
//   A (16-bit 16x32): lane<16 row l K={kb+0..7, kb+8..15}; lane>=16 K={16..23, 24..31}
//     -> two contiguous 16B runs (v8bf each) in element order.
//   B (32x16): lane<16 col l K=kb+0..15; lane>=16 K=kb+16..31
//     -> one contiguous 32B run of BT in element order.
//   C/D f32: VGPR r -> row r (+8 for lanes>=16), col = lane&15.
// ---------------------------------------------------------------------------
__global__ void k_gemm_wmma(const __bf16* __restrict__ A, const __bf16* __restrict__ BT,
                            float* __restrict__ C, int N, int Kdim, int Mdim) {
  const int wavesPerBlock = blockDim.x >> 5;
  const int waveId = threadIdx.x >> 5;
  const int lane   = threadIdx.x & 31;
  const int tileId = blockIdx.x * wavesPerBlock + waveId;
  const int stripsM = Mdim >> 5;                // 32 cols per wave
  const int tileRow = tileId / stripsM;
  const int strip   = tileId - tileRow * stripsM;
  if (tileRow * 16 >= N) return;                // wave-uniform guard

  const int row0 = tileRow * 16;
  const int col0 = strip * 32;
  const bool hi  = lane >= 16;
  const int  l   = lane & 15;

  const __bf16* arow = A  + (size_t)(row0 + l) * Kdim;
  const __bf16* bt0  = BT + (size_t)(col0 + l) * Kdim;
  const __bf16* bt1  = BT + (size_t)(col0 + 16 + l) * Kdim;
  const int aoff0 = hi ? 16 : 0;
  const int aoff1 = hi ? 24 : 8;
  const int boff  = hi ? 16 : 0;

  v8f acc0 = {}, acc1 = {};
#pragma unroll 4
  for (int kb = 0; kb < Kdim; kb += 32) {
    v8bf a0 = *(const v8bf*)(arow + kb + aoff0);
    v8bf a1 = *(const v8bf*)(arow + kb + aoff1);
    v16bf a = __builtin_shufflevector(a0, a1, 0, 1, 2, 3, 4, 5, 6, 7,
                                      8, 9, 10, 11, 12, 13, 14, 15);
    v16bf b0 = *(const v16bf*)(bt0 + kb + boff);
    v16bf b1 = *(const v16bf*)(bt1 + kb + boff);
    acc0 = __builtin_amdgcn_wmma_f32_16x16x32_bf16(false, a, false, b0,
                                                   (short)0, acc0, false, false);
    acc1 = __builtin_amdgcn_wmma_f32_16x16x32_bf16(false, a, false, b1,
                                                   (short)0, acc1, false, false);
  }
  const int crow = row0 + (hi ? 8 : 0);
#pragma unroll
  for (int r = 0; r < 8; ++r) {
    C[(size_t)(crow + r) * Mdim + col0 + l]      = acc0[r];
    C[(size_t)(crow + r) * Mdim + col0 + 16 + l] = acc1[r];
  }
}

// ---------------------------------------------------------------------------
// Substrate layer-1 projection: z[n, c] = sum_{k<4} feat[n,k] * W[k,c]  (K=4)
// ---------------------------------------------------------------------------
__global__ void k_lin4(const float* __restrict__ feat, const float* __restrict__ W,
                       float* __restrict__ z, int N, int M) {
  int idx = blockIdx.x * blockDim.x + threadIdx.x;
  if (idx >= N * M) return;
  int n = idx / M, c = idx - n * M;
  const float* f = feat + (size_t)n * 4;
  z[idx] = f[0] * W[c] + f[1] * W[M + c] + f[2] * W[2 * M + c] + f[3] * W[3 * M + c];
}

// attention scores: el/er[n,h] = sum_d z[n,h,d] * a_{l,r}[h,d]
__global__ void k_scores(const float* __restrict__ z, const float* __restrict__ al,
                         const float* __restrict__ ar, float* __restrict__ el,
                         float* __restrict__ er, int N, int H, int D) {
  int idx = blockIdx.x * blockDim.x + threadIdx.x;
  if (idx >= N * H) return;
  int h = idx % H;
  const float* zp = z + (size_t)idx * D;
  float sl = 0.f, sr = 0.f;
  for (int d = 0; d < D; ++d) { float v = zp[d]; sl += v * al[h * D + d]; sr += v * ar[h * D + d]; }
  el[idx] = sl; er[idx] = sr;
}

__global__ void k_fill(float* __restrict__ p, float v, int n) {
  int i = blockIdx.x * blockDim.x + threadIdx.x;
  if (i < n) p[i] = v;
}

DEVI float leaky(float v) { return v > 0.f ? v : 0.2f * v; }

__global__ void k_edge_max(const int* __restrict__ src, const int* __restrict__ dst,
                           const float* __restrict__ el, const float* __restrict__ er,
                           float* __restrict__ m, int E, int H) {
  int idx = blockIdx.x * blockDim.x + threadIdx.x;
  if (idx >= E * H) return;
  int e = idx / H, h = idx - e * H;
  int s = src[e], d = dst[e];
  atomMaxF(&m[d * H + h], leaky(el[s * H + h] + er[d * H + h]));
}

__global__ void k_edge_sum(const int* __restrict__ src, const int* __restrict__ dst,
                           const float* __restrict__ el, const float* __restrict__ er,
                           const float* __restrict__ m, float* __restrict__ dn, int E, int H) {
  int idx = blockIdx.x * blockDim.x + threadIdx.x;
  if (idx >= E * H) return;
  int e = idx / H, h = idx - e * H;
  int s = src[e], d = dst[e];
  float v = leaky(el[s * H + h] + er[d * H + h]);
  atomAddF(&dn[d * H + h], __expf(v - m[d * H + h]));
}

__global__ void k_edge_agg(const int* __restrict__ src, const int* __restrict__ dst,
                           const float* __restrict__ el, const float* __restrict__ er,
                           const float* __restrict__ m, const float* __restrict__ dn,
                           const float* __restrict__ z, float* __restrict__ out,
                           int E, int H, int D) {
  int idx = blockIdx.x * blockDim.x + threadIdx.x;
  if (idx >= E * H) return;
  int e = idx / H, h = idx - e * H;
  int s = src[e], d = dst[e];
  int dh = d * H + h;
  float v = leaky(el[s * H + h] + er[dh]);
  float alpha = __expf(v - m[dh]) / dn[dh];
  const float* zs = z + (size_t)(s * H + h) * D;
  float* od = out + (size_t)dh * D;
  for (int i = 0; i < D; ++i) atomAddF(&od[i], alpha * zs[i]);
}

__global__ void k_finalize(float* __restrict__ out, const float* __restrict__ b,
                           int total, int width, int elu) {
  int i = blockIdx.x * blockDim.x + threadIdx.x;
  if (i >= total) return;
  float x = out[i] + b[i % width];
  if (elu) x = x > 0.f ? x : (__expf(x) - 1.f);
  out[i] = x;
}

// column mean over nodes: hp[d] = mean_n h[n, d]
__global__ void k_colmean(const float* __restrict__ h, float* __restrict__ hp, int N, int D) {
  __shared__ float red[256];
  int d = blockIdx.x;
  float s = 0.f;
  for (int n = threadIdx.x; n < N; n += blockDim.x) s += h[(size_t)n * D + d];
  red[threadIdx.x] = s; __syncthreads();
  for (int off = 128; off > 0; off >>= 1) {
    if (threadIdx.x < (unsigned)off) red[threadIdx.x] += red[threadIdx.x + off];
    __syncthreads();
  }
  if (threadIdx.x == 0) hp[d] = red[0] / (float)N;
}

// ---------------------------------------------------------------------------
// VNR GAT: one workgroup per VNR graph (10 nodes, 20 edges), all in LDS.
// ---------------------------------------------------------------------------
__device__ void vnr_layer(float* h, float* z, float* el, float* er, float* m, float* dn,
                          const int* es, const int* ed,
                          const float* W, const float* al, const float* ar, const float* b,
                          int inDim, int H, int D, bool elu_act) {
  const int NVn = 10, EVn = 20;
  int t = threadIdx.x, bs = blockDim.x;
  int outW = H * D;
  for (int o = t; o < NVn * outW; o += bs) {           // z = h @ W
    int n = o / outW, c = o - n * outW;
    float s = 0.f;
    for (int k = 0; k < inDim; ++k) s += h[n * inDim + k] * W[k * outW + c];
    z[o] = s;
  }
  __syncthreads();
  for (int o = t; o < NVn * H; o += bs) {              // el, er, init m/dn
    int n = o / H, hh = o - n * H;
    float sl = 0.f, sr = 0.f;
    for (int d = 0; d < D; ++d) {
      float v = z[(n * H + hh) * D + d];
      sl += v * al[hh * D + d]; sr += v * ar[hh * D + d];
    }
    el[o] = sl; er[o] = sr; m[o] = -1e30f; dn[o] = 0.f;
  }
  __syncthreads();
  for (int o = t; o < NVn * outW; o += bs) h[o] = 0.f; // h becomes agg target
  __syncthreads();
  for (int o = t; o < EVn * H; o += bs) {              // segment max
    int e = o / H, hh = o - e * H;
    float v = leaky(el[es[e] * H + hh] + er[ed[e] * H + hh]);
    float* addr = &m[ed[e] * H + hh];
    if (v >= 0.f) atomicMax((int*)addr, __float_as_int(v));
    else          atomicMin((unsigned int*)addr, __float_as_uint(v));
  }
  __syncthreads();
  for (int o = t; o < EVn * H; o += bs) {              // segment sum of exp
    int e = o / H, hh = o - e * H;
    float v = leaky(el[es[e] * H + hh] + er[ed[e] * H + hh]);
    atomicAdd(&dn[ed[e] * H + hh], __expf(v - m[ed[e] * H + hh]));
  }
  __syncthreads();
  for (int o = t; o < EVn * H * D; o += bs) {          // aggregate
    int e = o / (H * D); int r = o - e * (H * D); int hh = r / D; int d = r - hh * D;
    int dh = ed[e] * H + hh;
    float v = leaky(el[es[e] * H + hh] + er[dh]);
    float alpha = __expf(v - m[dh]) / dn[dh];
    atomicAdd(&h[dh * D + d], alpha * z[(es[e] * H + hh) * D + d]);
  }
  __syncthreads();
  for (int o = t; o < NVn * outW; o += bs) {           // bias + activation
    float x = h[o] + b[o % outW];
    if (elu_act) x = x > 0.f ? x : (__expf(x) - 1.f);
    h[o] = x;
  }
  __syncthreads();
}

__global__ void k_vnr(const float* __restrict__ feat, const int* __restrict__ src,
                      const int* __restrict__ dst,
                      const float* W0, const float* al0, const float* ar0, const float* b0,
                      const float* W1, const float* al1, const float* ar1, const float* b1,
                      const float* W2, const float* al2, const float* ar2, const float* b2,
                      float* __restrict__ h_stack) {
  __shared__ float h[10 * 256];
  __shared__ float z[10 * 256];
  __shared__ float el[40], er[40], m[40], dn[40];
  int k = blockIdx.x, t = threadIdx.x;
  const float* f = feat + (size_t)k * 10 * 3;
  const int* es = src + k * 20;
  const int* ed = dst + k * 20;
  for (int o = t; o < 30; o += blockDim.x) h[o] = f[o];
  __syncthreads();
  vnr_layer(h, z, el, er, m, dn, es, ed, W0, al0, ar0, b0, 3,   4, 64, true);
  vnr_layer(h, z, el, er, m, dn, es, ed, W1, al1, ar1, b1, 256, 4, 64, true);
  vnr_layer(h, z, el, er, m, dn, es, ed, W2, al2, ar2, b2, 256, 1, 64, false);
  for (int d = t; d < 64; d += blockDim.x) {           // max-pool over nodes
    float mx = h[d];
    for (int n = 1; n < 10; ++n) mx = fmaxf(mx, h[n * 64 + d]);
    h_stack[k * 64 + d] = mx;
  }
}

// ---------------------------------------------------------------------------
// Tail: ctx MLP, keys, GRU cell, pointer energies, critic. One block.
// ---------------------------------------------------------------------------
__global__ void k_tail(const float* __restrict__ hp, const float* __restrict__ h_stack,
                       const unsigned char* __restrict__ mask,
                       const float* cW0, const float* cb0, const float* cW1, const float* cb1,
                       const float* kW, const float* kb,
                       const float* Wih, const float* Whh, const float* bih, const float* bhh,
                       const float* dW1, const float* dW2, const float* dv,
                       const float* start_tok,
                       const float* crW0, const float* crb0, const float* crW1,
                       const float* crb1, const float* crW2, const float* crb2,
                       float* __restrict__ out) {
  __shared__ float x[192], t1[256], ctx[256];
  __shared__ float keys[20 * 256];
  __shared__ float gi[768], gh[768], hn[256], u[256];
  __shared__ float v1[128], v2[64], red[256];
  int t = threadIdx.x;

  if (t < 128) x[t] = hp[t];
  if (t < 64) {                                        // masked queue mean
    float s = 0.f; int cnt = 0;
    for (int k = 0; k < 20; ++k)
      if (mask[k]) { s += h_stack[k * 64 + t]; ++cnt; }
    x[128 + t] = cnt > 0 ? s / fmaxf((float)cnt, 1.f) : 0.f;
  }
  __syncthreads();
  { float s = cb0[t]; for (int i = 0; i < 192; ++i) s += x[i] * cW0[i * 256 + t]; t1[t] = fmaxf(s, 0.f); }
  __syncthreads();
  { float s = cb1[t]; for (int i = 0; i < 256; ++i) s += t1[i] * cW1[i * 256 + t]; ctx[t] = fmaxf(s, 0.f); }
  __syncthreads();
  for (int o = t; o < 20 * 256; o += 256) {            // keys = h_stack @ kW + kb
    int k = o >> 8, j = o & 255;
    float s = kb[j];
    for (int i = 0; i < 64; ++i) s += h_stack[k * 64 + i] * kW[i * 256 + j];
    keys[o] = s;
  }
  for (int j = t; j < 768; j += 256) {                 // gi = start @ Wih + bih
    float s = bih[j];
    for (int i = 0; i < 256; ++i) s += start_tok[i] * Wih[i * 768 + j];
    gi[j] = s;
  }
  __syncthreads();
  for (int j = t; j < 768; j += 256) {                 // gh = ctx @ Whh + bhh
    float s = bhh[j];
    for (int i = 0; i < 256; ++i) s += ctx[i] * Whh[i * 768 + j];
    gh[j] = s;
  }
  __syncthreads();
  {                                                    // GRU gates
    float r  = 1.f / (1.f + __expf(-(gi[t] + gh[t])));
    float zg = 1.f / (1.f + __expf(-(gi[256 + t] + gh[256 + t])));
    float ng = tanhf(gi[512 + t] + r * gh[512 + t]);
    hn[t] = (1.f - zg) * ng + zg * ctx[t];
  }
  __syncthreads();
  { float s = 0.f; for (int i = 0; i < 256; ++i) s += hn[i] * dW2[i * 256 + t]; u[t] = s; }
  __syncthreads();
  for (int k = 0; k < 20; ++k) {                       // pointer energies
    float s0 = u[t];
    for (int i = 0; i < 256; ++i) s0 += keys[k * 256 + i] * dW1[i * 256 + t];
    red[t] = tanhf(s0) * dv[t];
    __syncthreads();
    for (int off = 128; off > 0; off >>= 1) {
      if (t < off) red[t] += red[t + off];
      __syncthreads();
    }
    if (t == 0) out[k] = mask[k] ? red[0] : -1e9f;
    __syncthreads();
  }
  if (t < 128) { float s = crb0[t]; for (int i = 0; i < 192; ++i) s += x[i] * crW0[i * 128 + t]; v1[t] = fmaxf(s, 0.f); }
  __syncthreads();
  if (t < 64)  { float s = crb1[t]; for (int i = 0; i < 128; ++i) s += v1[i] * crW1[i * 64 + t]; v2[t] = fmaxf(s, 0.f); }
  __syncthreads();
  if (t == 0)  { float s = crb2[0]; for (int i = 0; i < 64; ++i) s += v2[i] * crW2[i]; out[20] = s; }
  out[21 + t] = hn[t];
}

// ---------------------------------------------------------------------------
extern "C" void kernel_launch(void* const* d_in, const int* in_sizes, int n_in,
                              void* d_out, int out_size, void* d_ws, size_t ws_size,
                              hipStream_t stream) {
  (void)in_sizes; (void)n_in; (void)out_size; (void)ws_size;
  const int N = 20000, E = 320000;

  const float* sub_feat = (const float*)d_in[0];
  const int*   sub_src  = (const int*)d_in[1];
  const int*   sub_dst  = (const int*)d_in[2];
  const float* vnr_feat = (const float*)d_in[3];
  const int*   vnr_src  = (const int*)d_in[4];
  const int*   vnr_dst  = (const int*)d_in[5];
  const unsigned char* mask = (const unsigned char*)d_in[6];
  // sub_gat (3 layers x (W, a_l, a_r, b))
  const float* sW0 = (const float*)d_in[7],  *sal0 = (const float*)d_in[8],
             * sar0 = (const float*)d_in[9], *sb0 = (const float*)d_in[10];
  const float* sW1 = (const float*)d_in[11], *sal1 = (const float*)d_in[12],
             * sar1 = (const float*)d_in[13], *sb1 = (const float*)d_in[14];
  const float* sW2 = (const float*)d_in[15], *sal2 = (const float*)d_in[16],
             * sar2 = (const float*)d_in[17], *sb2 = (const float*)d_in[18];
  // vnr_gat
  const float* vW0 = (const float*)d_in[19], *val0 = (const float*)d_in[20],
             * var0 = (const float*)d_in[21], *vb0 = (const float*)d_in[22];
  const float* vW1 = (const float*)d_in[23], *val1 = (const float*)d_in[24],
             * var1 = (const float*)d_in[25], *vb1 = (const float*)d_in[26];
  const float* vW2 = (const float*)d_in[27], *val2 = (const float*)d_in[28],
             * var2 = (const float*)d_in[29], *vb2 = (const float*)d_in[30];
  const float* cW0 = (const float*)d_in[31], *cb0 = (const float*)d_in[32],
             * cW1 = (const float*)d_in[33], *cb1 = (const float*)d_in[34];
  const float* kW = (const float*)d_in[35], *kb = (const float*)d_in[36];
  const float* Wih = (const float*)d_in[37], *Whh = (const float*)d_in[38],
             * bih = (const float*)d_in[39], *bhh = (const float*)d_in[40],
             * dW1 = (const float*)d_in[41], *dW2 = (const float*)d_in[42],
             * dv  = (const float*)d_in[43];
  const float* start_tok = (const float*)d_in[44];
  const float* crW0 = (const float*)d_in[45], *crb0 = (const float*)d_in[46],
             * crW1 = (const float*)d_in[47], *crb1 = (const float*)d_in[48],
             * crW2 = (const float*)d_in[49], *crb2 = (const float*)d_in[50];
  float* out = (float*)d_out;

  // carve workspace (~105 MB; working set fits in 192 MB L2 -> atomics and
  // GEMM re-reads are L2-resident; HBM sees only compulsory traffic)
  char* w = (char*)d_ws;
  auto carve = [&](size_t bytes) {
    void* p = (void*)w; w += (bytes + 255) & ~(size_t)255; return p;
  };
  float*  bufH = (float*)carve((size_t)N * 512 * 4);
  float*  bufZ = (float*)carve((size_t)N * 512 * 4);
  __bf16* a16  = (__bf16*)carve((size_t)N * 512 * 2);
  __bf16* wt16 = (__bf16*)carve((size_t)512 * 512 * 2);
  float*  el   = (float*)carve((size_t)N * 4 * 4);
  float*  er   = (float*)carve((size_t)N * 4 * 4);
  float*  m    = (float*)carve((size_t)N * 4 * 4);
  float*  dn   = (float*)carve((size_t)N * 4 * 4);
  float*  hp   = (float*)carve(128 * 4);
  float*  hstk = (float*)carve(20 * 64 * 4);

  auto nb = [](long n) { return (unsigned)((n + 255) / 256); };

  // ---- substrate layer 0 (K=4 projection) ----
  k_lin4<<<nb((long)N * 512), 256, 0, stream>>>(sub_feat, sW0, bufZ, N, 512);
  k_scores<<<nb((long)N * 4), 256, 0, stream>>>(bufZ, sal0, sar0, el, er, N, 4, 128);
  k_fill<<<nb((long)N * 4), 256, 0, stream>>>(m, -1e30f, N * 4);
  k_fill<<<nb((long)N * 4), 256, 0, stream>>>(dn, 0.f, N * 4);
  k_fill<<<nb((long)N * 512), 256, 0, stream>>>(bufH, 0.f, N * 512);
  k_edge_max<<<nb((long)E * 4), 256, 0, stream>>>(sub_src, sub_dst, el, er, m, E, 4);
  k_edge_sum<<<nb((long)E * 4), 256, 0, stream>>>(sub_src, sub_dst, el, er, m, dn, E, 4);
  k_edge_agg<<<nb((long)E * 4), 256, 0, stream>>>(sub_src, sub_dst, el, er, m, dn, bufZ, bufH, E, 4, 128);
  k_finalize<<<nb((long)N * 512), 256, 0, stream>>>(bufH, sb0, N * 512, 512, 1);

  // ---- substrate layer 1: WMMA GEMM 20000x512 @ 512x512 (bf16 operands) ----
  k_cvt_bf16<<<nb((long)N * 512), 256, 0, stream>>>(bufH, a16, N * 512);
  k_transpose_bf16<<<nb(512L * 512), 256, 0, stream>>>(sW1, wt16, 512, 512);
  k_gemm_wmma<<<(N / 16) * (512 / 32) / 8, 256, 0, stream>>>(a16, wt16, bufZ, N, 512, 512);
  k_scores<<<nb((long)N * 4), 256, 0, stream>>>(bufZ, sal1, sar1, el, er, N, 4, 128);
  k_fill<<<nb((long)N * 4), 256, 0, stream>>>(m, -1e30f, N * 4);
  k_fill<<<nb((long)N * 4), 256, 0, stream>>>(dn, 0.f, N * 4);
  k_fill<<<nb((long)N * 512), 256, 0, stream>>>(bufH, 0.f, N * 512);
  k_edge_max<<<nb((long)E * 4), 256, 0, stream>>>(sub_src, sub_dst, el, er, m, E, 4);
  k_edge_sum<<<nb((long)E * 4), 256, 0, stream>>>(sub_src, sub_dst, el, er, m, dn, E, 4);
  k_edge_agg<<<nb((long)E * 4), 256, 0, stream>>>(sub_src, sub_dst, el, er, m, dn, bufZ, bufH, E, 4, 128);
  k_finalize<<<nb((long)N * 512), 256, 0, stream>>>(bufH, sb1, N * 512, 512, 1);

  // ---- substrate layer 2: WMMA GEMM 20000x512 @ 512x128, H=1 ----
  k_cvt_bf16<<<nb((long)N * 512), 256, 0, stream>>>(bufH, a16, N * 512);
  k_transpose_bf16<<<nb(512L * 128), 256, 0, stream>>>(sW2, wt16, 512, 128);
  k_gemm_wmma<<<(N / 16) * (128 / 32) / 8, 256, 0, stream>>>(a16, wt16, bufZ, N, 512, 128);
  k_scores<<<nb((long)N), 256, 0, stream>>>(bufZ, sal2, sar2, el, er, N, 1, 128);
  k_fill<<<nb((long)N), 256, 0, stream>>>(m, -1e30f, N);
  k_fill<<<nb((long)N), 256, 0, stream>>>(dn, 0.f, N);
  k_fill<<<nb((long)N * 128), 256, 0, stream>>>(bufH, 0.f, N * 128);
  k_edge_max<<<nb((long)E), 256, 0, stream>>>(sub_src, sub_dst, el, er, m, E, 1);
  k_edge_sum<<<nb((long)E), 256, 0, stream>>>(sub_src, sub_dst, el, er, m, dn, E, 1);
  k_edge_agg<<<nb((long)E), 256, 0, stream>>>(sub_src, sub_dst, el, er, m, dn, bufZ, bufH, E, 1, 128);
  k_finalize<<<nb((long)N * 128), 256, 0, stream>>>(bufH, sb2, N * 128, 128, 0);
  k_colmean<<<128, 256, 0, stream>>>(bufH, hp, N, 128);

  // ---- VNR encoders (one workgroup per graph, LDS-resident) ----
  k_vnr<<<20, 256, 0, stream>>>(vnr_feat, vnr_src, vnr_dst,
                                vW0, val0, var0, vb0, vW1, val1, var1, vb1,
                                vW2, val2, var2, vb2, hstk);

  // ---- heads: ctx MLP, keys, GRU, pointer energies, critic ----
  k_tail<<<1, 256, 0, stream>>>(hp, hstk, mask, cW0, cb0, cW1, cb1, kW, kb,
                                Wih, Whh, bih, bhh, dW1, dW2, dv, start_tok,
                                crW0, crb0, crW1, crb1, crW2, crb2, out);
}
